// TinyTransformer_42571715838596
// MI455X (gfx1250) — compile-verified, matrix operand order checked by reading, not run on hardware
//
#include <hip/hip_runtime.h>
#include <stdint.h>

// ---------------------------------------------------------------- constants
static constexpr int Vv  = 32000;
static constexpr int Dd  = 1024;
static constexpr int Ss  = 2048;
static constexpr int Hh  = 16;
static constexpr int Bb  = 4;
static constexpr int DFF = 4096;
static constexpr int HD  = 64;     // Dd / Hh
static constexpr int Mtot = Bb * Ss;   // 8192 rows of activations

typedef __bf16 bf16_t;
typedef __attribute__((ext_vector_type(8)))  bf16_t v8bf;
typedef __attribute__((ext_vector_type(16))) bf16_t v16bf;
typedef __attribute__((ext_vector_type(8)))  float  v8f;

// ---- gfx1250 async global->LDS copy path (probe via __has_builtin) --------
#if defined(__has_builtin)
#  if __has_builtin(__builtin_amdgcn_global_load_async_to_lds_b128)
#    define HAVE_ASYNC_LDS 1
#  else
#    define HAVE_ASYNC_LDS 0
#  endif
#  if __has_builtin(__builtin_amdgcn_s_wait_asynccnt)
#    define HAVE_WAIT_ASYNC 1
#  else
#    define HAVE_WAIT_ASYNC 0
#  endif
#else
#  define HAVE_ASYNC_LDS 0
#  define HAVE_WAIT_ASYNC 0
#endif

#if HAVE_ASYNC_LDS
// The builtin's params are AS1/AS3 pointers to a GCC-style int4 vector
// (diagnostic-confirmed: "int __attribute__((vector_size(16))) __device__*").
// Route casts through uintptr_t: generic->AS1 is value-preserving, and the
// low 32 bits of a generic LDS pointer are the LDS offset on CDNA5.
typedef int v4i_g __attribute__((vector_size(16)));
typedef __attribute__((address_space(1))) v4i_g* as1_v4i;
typedef __attribute__((address_space(3))) v4i_g* as3_v4i;
#define GLDS_B128(gp, lp)                                 \
  __builtin_amdgcn_global_load_async_to_lds_b128(         \
      (as1_v4i)(uintptr_t)(gp),                           \
      (as3_v4i)(uint32_t)(uintptr_t)(lp), 0, 0)
#if HAVE_WAIT_ASYNC
#define WAIT_ASYNC(n) __builtin_amdgcn_s_wait_asynccnt(n)
#else
#define WAIT_ASYNC(n) asm volatile("s_wait_asynccnt %0" ::"i"(n) : "memory")
#endif
#endif  // HAVE_ASYNC_LDS

// Load one 16-bit WMMA fragment slice for this lane.
// Per ISA layout, lane (r = lane&15, g = lane>>4) holds K = {8g..8g+7} and
// {16+8g..16+8g+7}; caller passes base = row_ptr + k_base + 8*g, so we read
// base[0..7] and base[16..23] (two 16B loads, works for global or LDS).
__device__ __forceinline__ v16bf ldfrag(const bf16_t* base) {
  v8bf lo = *(const v8bf*)(base);
  v8bf hi = *(const v8bf*)(base + 16);
  return __builtin_shufflevector(lo, hi, 0,1,2,3,4,5,6,7,8,9,10,11,12,13,14,15);
}

__device__ __forceinline__ v8f wmma_bf16(v16bf a, v16bf b, v8f c) {
  return __builtin_amdgcn_wmma_f32_16x16x32_bf16(false, a, false, b,
                                                 (short)0, c, false, false);
}

// ---------------------------------------------------------------- f32 -> bf16
__global__ __launch_bounds__(256) void cvt_bf16_kernel(
    const float* __restrict__ in, bf16_t* __restrict__ out, long n) {
  long i = (long)blockIdx.x * blockDim.x + threadIdx.x;
  if (i < n) out[i] = (bf16_t)in[i];
}

// ---------------------------------------------------------------- embedding
__global__ __launch_bounds__(256) void embed_kernel(
    const int* __restrict__ tokens, const float* __restrict__ emb,
    const float* __restrict__ pos, float* __restrict__ xf,
    bf16_t* __restrict__ xb) {
  long idx = (long)blockIdx.x * blockDim.x + threadIdx.x;
  if (idx >= (long)Mtot * Dd) return;
  int d = (int)(idx % Dd);
  long bs = idx / Dd;
  int s = (int)(bs % Ss);
  int tok = tokens[bs];
  float v = emb[(long)tok * Dd + d] + pos[(long)s * Dd + d];
  xf[idx] = v;
  xb[idx] = (bf16_t)v;
}

// ---------------------------------------------------------------- GEMM (NT)
// C[m,n] = sum_k A[m,k] * W[n,k] + bias[n]   (A: MxK, W: NxK, both bf16)
// block = 256 threads (8 waves), block tile 128x128, wave tile 32x64.
// Async path: 128x32 A/B tiles double-buffered in LDS via
// global_load_async_to_lds_b128, fragments read with ds_load_b128.
__global__ __launch_bounds__(256) void gemm_nt_kernel(
    const bf16_t* __restrict__ A, const bf16_t* __restrict__ W,
    const float* __restrict__ bias, float* __restrict__ Cf,
    bf16_t* __restrict__ Cb, int M, int N, int K, int relu) {
  const int lane = threadIdx.x & 31;
  const int wave = threadIdx.x >> 5;
  const int r = lane & 15, g = lane >> 4;
  const int mblk = blockIdx.y * 128;
  const int nblk = blockIdx.x * 128;
  const int wm = wave & 3, wn = wave >> 2;
  const int m0 = mblk + wm * 32;  // wave rows [m0, m0+32)
  const int n0 = nblk + wn * 64;  // wave cols [n0, n0+64)

  v8f acc[2][4] = {};

#if HAVE_ASYNC_LDS
  __shared__ __align__(16) bf16_t Abuf[2][128 * 32];
  __shared__ __align__(16) bf16_t Bbuf[2][128 * 32];
  const int tid = threadIdx.x;

  // each thread copies 2 x 16B chunks of A and of B per 128x32 tile
  auto issue_tile = [&](int buf, int kb) {
#pragma unroll
    for (int i = 0; i < 2; ++i) {
      const int li = tid + i * 256;        // 0..511
      const int row = li >> 2;             // 0..127
      const int cg = (li & 3) * 8;         // col offset 0/8/16/24
      GLDS_B128(A + (size_t)(mblk + row) * K + kb + cg,
                &Abuf[buf][row * 32 + cg]);
      GLDS_B128(W + (size_t)(nblk + row) * K + kb + cg,
                &Bbuf[buf][row * 32 + cg]);
    }
  };

  issue_tile(0, 0);
  for (int kb = 0; kb < K; kb += 32) {
    const int cur = (kb >> 5) & 1;
    if (kb + 32 < K) {
      issue_tile(cur ^ 1, kb + 32);  // overlap next tile with this one's math
      WAIT_ASYNC(4);                 // 4 outstanding = next tile in flight
    } else {
      WAIT_ASYNC(0);
    }
    __syncthreads();  // all waves' tile data visible in LDS

    v16bf a[2];
#pragma unroll
    for (int i = 0; i < 2; ++i)
      a[i] = ldfrag(&Abuf[cur][(wm * 32 + i * 16 + r) * 32 + g * 8]);
    v16bf b[4];
#pragma unroll
    for (int t = 0; t < 4; ++t)
      b[t] = ldfrag(&Bbuf[cur][(wn * 64 + t * 16 + r) * 32 + g * 8]);
#pragma unroll
    for (int i = 0; i < 2; ++i)
#pragma unroll
      for (int t = 0; t < 4; ++t)
        acc[i][t] = wmma_bf16(a[i], b[t], acc[i][t]);

    __syncthreads();  // protect buffer before it is re-filled
  }
#else
  // fallback: direct global loads (L2 absorbs intra-block reuse)
  const bf16_t* arow0 = A + (size_t)(m0 + r) * K + g * 8;
  const bf16_t* arow1 = A + (size_t)(m0 + 16 + r) * K + g * 8;
  for (int kb = 0; kb < K; kb += 32) {
    if (kb + 32 < K) {
      __builtin_prefetch(arow0 + kb + 32, 0, 3);
      __builtin_prefetch(W + (size_t)(n0 + r) * K + kb + 32 + g * 8, 0, 3);
    }
    v16bf a[2];
    a[0] = ldfrag(arow0 + kb);
    a[1] = ldfrag(arow1 + kb);
    v16bf b[4];
#pragma unroll
    for (int t = 0; t < 4; ++t)
      b[t] = ldfrag(W + (size_t)(n0 + t * 16 + r) * K + kb + g * 8);
#pragma unroll
    for (int i = 0; i < 2; ++i)
#pragma unroll
      for (int t = 0; t < 4; ++t)
        acc[i][t] = wmma_bf16(a[i], b[t], acc[i][t]);
  }
#endif

  // Epilogue: C/D layout -> VGPR e holds row (e + 8g), lane col r.
  // f32 outputs use non-temporal stores so the 1GB logits stream does not
  // evict bf16 w_fc (66MB) from the 192MB L2.
#pragma unroll
  for (int i = 0; i < 2; ++i) {
#pragma unroll
    for (int t = 0; t < 4; ++t) {
      const int col = n0 + t * 16 + r;
      const float bv = bias ? bias[col] : 0.0f;
#pragma unroll
      for (int e = 0; e < 8; ++e) {
        const int row = m0 + i * 16 + e + 8 * g;
        float v = acc[i][t][e] + bv;
        if (relu) v = fmaxf(v, 0.0f);
        const size_t o = (size_t)row * N + col;
        if (Cf) __builtin_nontemporal_store(v, &Cf[o]);
        if (Cb) Cb[o] = (bf16_t)v;
      }
    }
  }
}

// ---------------------------------------------------------------- QKV split
// qkv f32 [B,S,3D]  ->  q,k bf16 [B,H,S,HD],  v transposed bf16 [B,H,HD,S]
__global__ __launch_bounds__(256) void qkv_split_kernel(
    const float* __restrict__ qkv, bf16_t* __restrict__ q,
    bf16_t* __restrict__ k, bf16_t* __restrict__ vt) {
  long idx = (long)blockIdx.x * blockDim.x + threadIdx.x;
  if (idx >= (long)Mtot * 3 * Dd) return;
  int e = (int)(idx % (3 * Dd));
  long bs = idx / (3 * Dd);
  int s = (int)(bs % Ss);
  int b = (int)(bs / Ss);
  int which = e / Dd;
  int c = e % Dd;
  int h = c / HD;
  int dd = c % HD;
  long bh = (long)b * Hh + h;
  bf16_t v = (bf16_t)qkv[idx];
  if (which == 0)      q[(bh * Ss + s) * HD + dd] = v;
  else if (which == 1) k[(bh * Ss + s) * HD + dd] = v;
  else                 vt[(bh * HD + dd) * Ss + s] = v;
}

// ---------------------------------------------------------------- attention
// Flash-style: one wave handles 16 query rows, streams keys in blocks of 32.
__global__ __launch_bounds__(32) void flash_attn_kernel(
    const bf16_t* __restrict__ Q, const bf16_t* __restrict__ Km,
    const bf16_t* __restrict__ Vt, bf16_t* __restrict__ Ctx) {
  __shared__ __align__(16) bf16_t pt[16 * 32];  // prob tile staging (LDS)

  const int lane = threadIdx.x & 31;
  const int r = lane & 15, g = lane >> 4;
  const int qt = blockIdx.x & (Ss / 16 - 1);  // 128 q-tiles
  const int bh = blockIdx.x >> 7;
  const int b = bh >> 4, h = bh & (Hh - 1);
  const int q0 = qt * 16;

  const bf16_t* qp = Q  + (size_t)bh * Ss * HD;
  const bf16_t* kp = Km + (size_t)bh * Ss * HD;
  const bf16_t* vp = Vt + (size_t)bh * HD * Ss;

  // Loop-invariant Q fragments (K-dim = HD = 64 -> two K=32 chunks)
  v16bf aq0 = ldfrag(qp + (size_t)(q0 + r) * HD + 0 + g * 8);
  v16bf aq1 = ldfrag(qp + (size_t)(q0 + r) * HD + 32 + g * 8);

  float mrun[8], lrun[8];
#pragma unroll
  for (int i = 0; i < 8; ++i) { mrun[i] = -1e30f; lrun[i] = 0.0f; }
  v8f acc[4] = {};  // ctx accumulators, 4 d-tiles of 16 -> HD=64

  const float sc = 0.125f;  // 1/sqrt(HD)

  for (int kb = 0; kb < Ss; kb += 32) {
    // scores for key cols [kb, kb+16) and [kb+16, kb+32)
    v8f s0 = {}, s1 = {};
    {
      v16bf b00 = ldfrag(kp + (size_t)(kb + r) * HD + 0 + g * 8);
      v16bf b01 = ldfrag(kp + (size_t)(kb + r) * HD + 32 + g * 8);
      s0 = wmma_bf16(aq0, b00, s0);
      s0 = wmma_bf16(aq1, b01, s0);
      v16bf b10 = ldfrag(kp + (size_t)(kb + 16 + r) * HD + 0 + g * 8);
      v16bf b11 = ldfrag(kp + (size_t)(kb + 16 + r) * HD + 32 + g * 8);
      s1 = wmma_bf16(aq0, b10, s1);
      s1 = wmma_bf16(aq1, b11, s1);
    }
    // online softmax per row (row = i + 8g; reduce across the 16 N-lanes —
    // xor masks <16 stay inside each wave half)
#pragma unroll
    for (int i = 0; i < 8; ++i) {
      float v0 = s0[i] * sc, v1 = s1[i] * sc;
      float tmax = fmaxf(v0, v1);
#pragma unroll
      for (int msk = 1; msk < 16; msk <<= 1)
        tmax = fmaxf(tmax, __shfl_xor(tmax, msk, 32));
      float mnew = fmaxf(mrun[i], tmax);
      float cf = __expf(mrun[i] - mnew);
      mrun[i] = mnew;
      float p0 = __expf(v0 - mnew);
      float p1 = __expf(v1 - mnew);
      float rs = p0 + p1;
#pragma unroll
      for (int msk = 1; msk < 16; msk <<= 1)
        rs += __shfl_xor(rs, msk, 32);
      lrun[i] = lrun[i] * cf + rs;
#pragma unroll
      for (int t = 0; t < 4; ++t) acc[t][i] *= cf;
      // stage prob tile to LDS for the cross-lane C->A transpose
      pt[(i + 8 * g) * 32 + r]      = (bf16_t)p0;
      pt[(i + 8 * g) * 32 + 16 + r] = (bf16_t)p1;
    }
    __syncthreads();  // order LDS write -> read (single wave; DS is in-order)

    v16bf ap = ldfrag(pt + r * 32 + g * 8);  // prob tile as A fragment
#pragma unroll
    for (int t = 0; t < 4; ++t) {
      v16bf bv = ldfrag(vp + (size_t)(t * 16 + r) * Ss + kb + g * 8);
      acc[t] = wmma_bf16(ap, bv, acc[t]);
    }
    __syncthreads();  // protect pt before next iteration overwrites
  }

  // finalize: divide by row sums, scatter into [B,S,D] bf16 layout
#pragma unroll
  for (int i = 0; i < 8; ++i) {
    const float inv = 1.0f / lrun[i];
    const int row = q0 + i + 8 * g;
    const size_t o = ((size_t)b * Ss + row) * Dd + h * HD;
#pragma unroll
    for (int t = 0; t < 4; ++t)
      Ctx[o + t * 16 + r] = (bf16_t)(acc[t][i] * inv);
  }
}

// ---------------------------------------------------------------- resid + LN
__global__ __launch_bounds__(256) void resid_ln_kernel(
    const float* __restrict__ X, const float* __restrict__ Y,
    const float* __restrict__ G, const float* __restrict__ Bt,
    float* __restrict__ Of, bf16_t* __restrict__ Ob) {
  __shared__ float red[256];
  const int row = blockIdx.x;
  const int t = threadIdx.x;
  const float* xr = X + (size_t)row * Dd;
  const float* yr = Y + (size_t)row * Dd;

  float s = 0.0f;
  for (int i = t; i < Dd; i += 256) s += xr[i] + yr[i];
  red[t] = s; __syncthreads();
  for (int w = 128; w > 0; w >>= 1) { if (t < w) red[t] += red[t + w]; __syncthreads(); }
  const float mu = red[0] * (1.0f / Dd);
  __syncthreads();

  float s2 = 0.0f;
  for (int i = t; i < Dd; i += 256) { float d = xr[i] + yr[i] - mu; s2 += d * d; }
  red[t] = s2; __syncthreads();
  for (int w = 128; w > 0; w >>= 1) { if (t < w) red[t] += red[t + w]; __syncthreads(); }
  const float rstd = rsqrtf(red[0] * (1.0f / Dd) + 1e-5f);

  for (int i = t; i < Dd; i += 256) {
    float v = (xr[i] + yr[i] - mu) * rstd * G[i] + Bt[i];
    size_t o = (size_t)row * Dd + i;
    if (Of) Of[o] = v;
    if (Ob) Ob[o] = (bf16_t)v;
  }
}

// ---------------------------------------------------------------- launcher
extern "C" void kernel_launch(void* const* d_in, const int* in_sizes, int n_in,
                              void* d_out, int out_size, void* d_ws,
                              size_t ws_size, hipStream_t stream) {
  (void)in_sizes; (void)n_in; (void)out_size; (void)ws_size;
  const int*   tokens = (const int*)  d_in[0];
  const float* emb    = (const float*)d_in[1];
  const float* pos    = (const float*)d_in[2];
  const float* w_qkv  = (const float*)d_in[3];
  const float* b_qkv  = (const float*)d_in[4];
  const float* w_o    = (const float*)d_in[5];
  const float* b_o    = (const float*)d_in[6];
  const float* g1     = (const float*)d_in[7];
  const float* beta1  = (const float*)d_in[8];
  const float* w1     = (const float*)d_in[9];
  const float* b1     = (const float*)d_in[10];
  const float* w2     = (const float*)d_in[11];
  const float* b2     = (const float*)d_in[12];
  const float* g2     = (const float*)d_in[13];
  const float* beta2  = (const float*)d_in[14];
  const float* w_fc   = (const float*)d_in[15];
  const float* b_fc   = (const float*)d_in[16];
  float* logits = (float*)d_out;

  // ---- workspace carve (256B aligned) ----
  char* base = (char*)d_ws;
  size_t off = 0;
  auto alloc = [&](size_t bytes) -> char* {
    off = (off + 255) & ~(size_t)255;
    char* p = base + off;
    off += bytes;
    return p;
  };
  bf16_t* wqkv_b = (bf16_t*)alloc((size_t)3 * Dd * Dd * 2);
  bf16_t* wo_b   = (bf16_t*)alloc((size_t)Dd * Dd * 2);
  bf16_t* w1_b   = (bf16_t*)alloc((size_t)DFF * Dd * 2);
  bf16_t* w2_b   = (bf16_t*)alloc((size_t)Dd * DFF * 2);
  bf16_t* wfc_b  = (bf16_t*)alloc((size_t)Vv * Dd * 2);
  float*  xf     = (float*) alloc((size_t)Mtot * Dd * 4);
  bf16_t* xb     = (bf16_t*)alloc((size_t)Mtot * Dd * 2);
  float*  qkvf   = (float*) alloc((size_t)Mtot * 3 * Dd * 4);
  bf16_t* qb     = (bf16_t*)alloc((size_t)Mtot * Dd * 2);
  bf16_t* kb_    = (bf16_t*)alloc((size_t)Mtot * Dd * 2);
  bf16_t* vtb    = (bf16_t*)alloc((size_t)Mtot * Dd * 2);
  bf16_t* ctxb   = (bf16_t*)alloc((size_t)Mtot * Dd * 2);
  float*  attnf  = (float*) alloc((size_t)Mtot * Dd * 4);
  float*  x1f    = (float*) alloc((size_t)Mtot * Dd * 4);
  bf16_t* x1b    = (bf16_t*)alloc((size_t)Mtot * Dd * 2);
  bf16_t* hb     = (bf16_t*)alloc((size_t)Mtot * DFF * 2);
  float*  ffnf   = (float*) alloc((size_t)Mtot * Dd * 4);
  bf16_t* x2b    = (bf16_t*)alloc((size_t)Mtot * Dd * 2);

  auto cvt = [&](const float* src, bf16_t* dst, long n) {
    cvt_bf16_kernel<<<(unsigned)((n + 255) / 256), 256, 0, stream>>>(src, dst, n);
  };
  auto gemm = [&](const bf16_t* A, const bf16_t* W, const float* bias,
                  float* Cf, bf16_t* Cb, int M, int N, int K, int relu) {
    dim3 grid(N / 128, M / 128);
    gemm_nt_kernel<<<grid, 256, 0, stream>>>(A, W, bias, Cf, Cb, M, N, K, relu);
  };

  // 1) weights -> bf16
  cvt(w_qkv, wqkv_b, (long)3 * Dd * Dd);
  cvt(w_o,   wo_b,   (long)Dd * Dd);
  cvt(w1,    w1_b,   (long)DFF * Dd);
  cvt(w2,    w2_b,   (long)Dd * DFF);
  cvt(w_fc,  wfc_b,  (long)Vv * Dd);

  // 2) x = emb[tokens] + pos
  {
    long n = (long)Mtot * Dd;
    embed_kernel<<<(unsigned)((n + 255) / 256), 256, 0, stream>>>(
        tokens, emb, pos, xf, xb);
  }

  // 3) qkv = x @ w_qkv^T + b_qkv
  gemm(xb, wqkv_b, b_qkv, qkvf, nullptr, Mtot, 3 * Dd, Dd, 0);

  // 4) split into q,k [B,H,S,HD] and v^T [B,H,HD,S]
  {
    long n = (long)Mtot * 3 * Dd;
    qkv_split_kernel<<<(unsigned)((n + 255) / 256), 256, 0, stream>>>(
        qkvf, qb, kb_, vtb);
  }

  // 5) flash attention -> ctx bf16 [B,S,D]
  flash_attn_kernel<<<Bb * Hh * (Ss / 16), 32, 0, stream>>>(qb, kb_, vtb, ctxb);

  // 6) attn_out = ctx @ w_o^T + b_o
  gemm(ctxb, wo_b, b_o, attnf, nullptr, Mtot, Dd, Dd, 0);

  // 7) x1 = LN(x + attn_out)
  resid_ln_kernel<<<Mtot, 256, 0, stream>>>(xf, attnf, g1, beta1, x1f, x1b);

  // 8) h = relu(x1 @ w1^T + b1)   (bf16 only)
  gemm(x1b, w1_b, b1, nullptr, hb, Mtot, DFF, Dd, 1);

  // 9) ffn = h @ w2^T + b2
  gemm(hb, w2_b, b2, ffnf, nullptr, Mtot, Dd, DFF, 0);

  // 10) x2 = LN(x1 + ffn)  (bf16 only, feeds final GEMM)
  resid_ln_kernel<<<Mtot, 256, 0, stream>>>(x1f, ffnf, g2, beta2, nullptr, x2b);

  // 11) logits = x2 @ w_fc^T + b_fc   (f32 out, 537 GFLOP, w_fc lives in L2)
  gemm(x2b, wfc_b, b_fc, logits, nullptr, Mtot, Vv, Dd, 0);
}
